// Gemma3EncoderAttention_30073361006886
// MI455X (gfx1250) — matrix operand order repaired and checked by reading, not compiled, not run on hardware
//
#include <hip/hip_runtime.h>
#include <hip/hip_bf16.h>

#define H 8
#define KV 4
#define DHEAD 256
#define HIDDEN 2560
#define SEQ 2048
#define BATCH 2
#define NQ (H*DHEAD)        // 2048
#define NKV (KV*DHEAD)      // 1024
#define NQKV (NQ + 2*NKV)   // 4096
#define MROWS (BATCH*SEQ)   // 4096
#define ATT_SCALE 0.0625f   // 256^-0.5

#if defined(__AMDGCN__) && __has_builtin(__builtin_amdgcn_global_load_async_to_lds_b128)
#define HAS_ASYNC_LDS 1
#else
#define HAS_ASYNC_LDS 0
#endif

typedef __attribute__((ext_vector_type(16))) __bf16 v16bf;
typedef __attribute__((ext_vector_type(8)))  __bf16 bf16x8;
typedef __attribute__((ext_vector_type(8)))  float  v8f;
typedef int v4i __attribute__((vector_size(16)));   // payload type for b128 async copies

static __device__ __forceinline__ v16bf cat8(bf16x8 lo, bf16x8 hi) {
  return __builtin_shufflevector(lo, hi, 0,1,2,3,4,5,6,7,8,9,10,11,12,13,14,15);
}
// A-fragment (16x32 bf16, lane m = lane&15): 8 elems at K=8*half, 8 at K=16+8*half
static __device__ __forceinline__ v16bf ldfragA(const __bf16* rowbase, int half) {
  const __bf16* p = rowbase + 8*half;
  return cat8(*(const bf16x8*)p, *(const bf16x8*)(p + 16));
}
// B-fragment (32x16 bf16, lane n = lane&15): 8 elems at K=8*half, 8 at K=16+8*half (from N-major storage)
static __device__ __forceinline__ v16bf ldfragB(const __bf16* rowbase, int half) {
  const __bf16* p = rowbase + 8*half;
  return cat8(*(const bf16x8*)p, *(const bf16x8*)(p + 16));
}
static __device__ __forceinline__ v8f wmma_bf16(v16bf a, v16bf b, v8f c) {
  return __builtin_amdgcn_wmma_f32_16x16x32_bf16(false, a, false, b, (short)0, c, false, false);
}

// 16-byte global -> LDS copy: async data mover when available (ASYNCcnt-tracked),
// otherwise VGPR round trip.
static __device__ __forceinline__ void async_cp16(const __bf16* g, __bf16* l) {
#if HAS_ASYNC_LDS
  __builtin_amdgcn_global_load_async_to_lds_b128(
      (__attribute__((address_space(1))) v4i*)(g),
      (__attribute__((address_space(3))) v4i*)(l), 0, 0);
#else
  *(bf16x8*)l = *(const bf16x8*)g;
#endif
}
static __device__ __forceinline__ void async_wait0() {
#if HAS_ASYNC_LDS
#if __has_builtin(__builtin_amdgcn_s_wait_asynccnt)
  __builtin_amdgcn_s_wait_asynccnt(0);
#else
  asm volatile("s_wait_asynccnt 0x0" ::: "memory");
#endif
#endif
}

// ---------------- conversion kernels ----------------
__global__ void cvt_f32_bf16_kernel(const float* __restrict__ src,
                                    __bf16* __restrict__ dst, int n) {
  int i = blockIdx.x*blockDim.x + threadIdx.x;
  if (i < n) dst[i] = (__bf16)src[i];
}

// fused transposed QKV weight: dst[n*HIDDEN + k], n in [0,4096)
__global__ void build_wqkvT_kernel(const float* __restrict__ wq,
                                   const float* __restrict__ wk,
                                   const float* __restrict__ wv,
                                   __bf16* __restrict__ dst) {
  int i = blockIdx.x*blockDim.x + threadIdx.x;
  int n = i / HIDDEN, k = i - n*HIDDEN;
  float v;
  if (n < NQ)            v = wq[(size_t)k*NQ + n];
  else if (n < NQ + NKV) v = wk[(size_t)k*NKV + (n - NQ)];
  else                   v = wv[(size_t)k*NKV + (n - NQ - NKV)];
  dst[i] = (__bf16)v;
}

// woT[n*NQ + k] = wo[k*HIDDEN + n], n in [0,2560), k in [0,2048)
__global__ void build_woT_kernel(const float* __restrict__ wo, __bf16* __restrict__ dst) {
  int i = blockIdx.x*blockDim.x + threadIdx.x;
  int n = i / NQ, k = i - n*NQ;
  dst[i] = (__bf16)wo[(size_t)k*HIDDEN + n];
}

// ---------------- bf16 WMMA GEMM: C[M][N](f32) = A[M][K] * Bt[N][K]^T ----------------
#define GP 40  // LDS row stride in bf16 (32 + 8 pad -> 80B, conflict-free b128 reads)
__global__ __launch_bounds__(256) void gemm_bf16_kernel(
    const __bf16* __restrict__ A, const __bf16* __restrict__ Bt,
    float* __restrict__ C, int K, int lda, int ldb, int ldc)
{
  __shared__ __align__(16) __bf16 As[2][128*GP];
  __shared__ __align__(16) __bf16 Bs[2][128*GP];
  const int tid  = threadIdx.x;
  const int wave = tid >> 5, lane = tid & 31;
  const int half = lane >> 4, lm = lane & 15;
  const int m0 = blockIdx.y * 128, n0 = blockIdx.x * 128;
  const int wm = (wave & 3) * 32, wn = (wave >> 2) * 64;
  const int lrow = tid >> 1, lk = (tid & 1) * 16;
  const __bf16* Ap = A  + (size_t)(m0 + lrow)*lda + lk;
  const __bf16* Bp = Bt + (size_t)(n0 + lrow)*ldb + lk;
  v8f acc[2][4] = {};

#if HAS_ASYNC_LDS
  // ---- double-buffered async pipeline: one barrier per K-step ----
  {
    __bf16* ad = &As[0][lrow*GP + lk];
    __bf16* bd = &Bs[0][lrow*GP + lk];
    async_cp16(Ap,     ad);  async_cp16(Ap + 8, ad + 8);
    async_cp16(Bp,     bd);  async_cp16(Bp + 8, bd + 8);
  }
  int cur = 0;
  for (int k0 = 0; k0 < K; k0 += 32) {
    async_wait0();       // this wave's tile writes (and any next-tile issue) complete
    __syncthreads();     // tile published; previous compute on other buffer retired
    if (k0 + 32 < K) {   // prefetch next tile into the other buffer while we compute
      __bf16* ad = &As[cur^1][lrow*GP + lk];
      __bf16* bd = &Bs[cur^1][lrow*GP + lk];
      async_cp16(Ap + k0 + 32, ad);  async_cp16(Ap + k0 + 40, ad + 8);
      async_cp16(Bp + k0 + 32, bd);  async_cp16(Bp + k0 + 40, bd + 8);
    }
    const __bf16* as = As[cur];
    const __bf16* bs = Bs[cur];
    v16bf af[2], bfr[4];
    #pragma unroll
    for (int t = 0; t < 2; ++t) af[t]  = ldfragA(&as[(wm + t*16 + lm)*GP], half);
    #pragma unroll
    for (int u = 0; u < 4; ++u) bfr[u] = ldfragB(&bs[(wn + u*16 + lm)*GP], half);
    #pragma unroll
    for (int t = 0; t < 2; ++t)
      #pragma unroll
      for (int u = 0; u < 4; ++u)
        acc[t][u] = wmma_bf16(af[t], bfr[u], acc[t][u]);
    cur ^= 1;
  }
#else
  // ---- fallback: VGPR-staged single buffer ----
  for (int k0 = 0; k0 < K; k0 += 32) {
    bf16x8 a0 = *(const bf16x8*)(Ap + k0);
    bf16x8 a1 = *(const bf16x8*)(Ap + k0 + 8);
    bf16x8 b0 = *(const bf16x8*)(Bp + k0);
    bf16x8 b1 = *(const bf16x8*)(Bp + k0 + 8);
    if (k0 + 32 < K) {
      __builtin_prefetch(Ap + k0 + 32, 0, 1);
      __builtin_prefetch(Bp + k0 + 32, 0, 1);
    }
    __syncthreads();
    *(bf16x8*)&As[0][lrow*GP + lk]     = a0;
    *(bf16x8*)&As[0][lrow*GP + lk + 8] = a1;
    *(bf16x8*)&Bs[0][lrow*GP + lk]     = b0;
    *(bf16x8*)&Bs[0][lrow*GP + lk + 8] = b1;
    __syncthreads();
    v16bf af[2], bfr[4];
    #pragma unroll
    for (int t = 0; t < 2; ++t) af[t]  = ldfragA(&As[0][(wm + t*16 + lm)*GP], half);
    #pragma unroll
    for (int u = 0; u < 4; ++u) bfr[u] = ldfragB(&Bs[0][(wn + u*16 + lm)*GP], half);
    #pragma unroll
    for (int t = 0; t < 2; ++t)
      #pragma unroll
      for (int u = 0; u < 4; ++u)
        acc[t][u] = wmma_bf16(af[t], bfr[u], acc[t][u]);
  }
#endif

  #pragma unroll
  for (int t = 0; t < 2; ++t)
    #pragma unroll
    for (int u = 0; u < 4; ++u)
      #pragma unroll
      for (int r = 0; r < 8; ++r)
        C[(size_t)(m0 + wm + t*16 + r + 8*half)*ldc + (n0 + wn + u*16 + lm)] = acc[t][u][r];
}

// ---------------- per-head RMSNorm + RoPE, f32 -> bf16 head layouts ----------------
__global__ __launch_bounds__(384) void rmsrope_kernel(
    const float* __restrict__ qkv, const float* __restrict__ qw,
    const float* __restrict__ kw, const int* __restrict__ pos_ids,
    __bf16* __restrict__ Qh, __bf16* __restrict__ Kh)
{
  const int row = blockIdx.x;                 // b*SEQ + s
  const int b = row / SEQ, s = row - b*SEQ;
  const int wave = threadIdx.x >> 5, lane = threadIdx.x & 31;
  const bool isq = wave < H;
  const int head = isq ? wave : wave - H;
  const float* src = qkv + (size_t)row*NQKV + (isq ? head*DHEAD : NQ + head*DHEAD);
  float v[8], ss = 0.f;
  #pragma unroll
  for (int j = 0; j < 8; ++j) { v[j] = src[j*32 + lane]; ss += v[j]*v[j]; }
  #pragma unroll
  for (int m = 16; m >= 1; m >>= 1) ss += __shfl_xor(ss, m, 32);
  const float rn = rsqrtf(ss * (1.f/256.f) + 1e-6f);
  const float* w = isq ? qw : kw;
  const float pos = (float)pos_ids[row];
  __bf16* dst = isq ? (Qh + ((size_t)(b*H  + head)*SEQ + s)*DHEAD)
                    : (Kh + ((size_t)(b*KV + head)*SEQ + s)*DHEAD);
  float nv[8];
  #pragma unroll
  for (int j = 0; j < 8; ++j) {
    int d = j*32 + lane;
    nv[j] = v[j]*rn*(1.f + w[d]);
  }
  #pragma unroll
  for (int j = 0; j < 8; ++j) {
    int d = j*32 + lane;
    int i = d & 127;                              // freq index
    float fr = pos * __expf(-0.1079336764f * (float)i);  // 1e6^(-i/128)
    float c = __cosf(fr), sn = __sinf(fr);
    float rot = (j < 4) ? -nv[j+4] : nv[j-4];     // rotate_half partner lives in same lane
    dst[d] = (__bf16)(nv[j]*c + rot*sn);
  }
}

// ---------------- V transpose: Vt[b][kv][d][s] bf16 ----------------
__global__ void vtrans_kernel(const float* __restrict__ qkv, __bf16* __restrict__ Vt) {
  int i = blockIdx.x*blockDim.x + threadIdx.x;    // ((b*KV+kv)*DHEAD + d)*SEQ + s
  int s = i & (SEQ-1);
  int rest = i >> 11;
  int d = rest & (DHEAD-1);
  int bk = rest >> 8;
  int b = bk >> 2, kvh = bk & 3;
  Vt[i] = (__bf16)qkv[((size_t)(b*SEQ + s))*NQKV + NQ + NKV + kvh*DHEAD + d];
}

// ---------------- causal flash attention (GQA 2:1), bf16 WMMA ----------------
#define KP 264  // 256+8 pad
#define VP 40   // 32+8 pad
__global__ __launch_bounds__(256) void attn_kernel(
    const __bf16* __restrict__ Qh, const __bf16* __restrict__ Kh,
    const __bf16* __restrict__ Vt, __bf16* __restrict__ Obf)
{
  __shared__ __align__(16) __bf16 Ks[32*KP];
  __shared__ __align__(16) __bf16 Vs[256*VP];
  __shared__ __align__(16) __bf16 Ps[8*16*VP];    // per-wave P scratch
  const int tid  = threadIdx.x;
  const int wave = tid >> 5, lane = tid & 31;
  const int half = lane >> 4, lm = lane & 15;
  const int h = blockIdx.y, b = blockIdx.z;
  const int kvh = h >> 1;                         // groups = H/KV = 2
  const int q0 = blockIdx.x*128 + wave*16;
  const __bf16* Qbase = Qh + ((size_t)(b*H + h)*SEQ + q0)*DHEAD;
  v16bf qf[8];
  #pragma unroll
  for (int kk = 0; kk < 8; ++kk)
    qf[kk] = ldfragA(Qbase + (size_t)lm*DHEAD + kk*32, half);
  v8f o[16] = {};
  float mst[8], lst[8];
  #pragma unroll
  for (int r = 0; r < 8; ++r) { mst[r] = -1e30f; lst[r] = 0.f; }
  const __bf16* Kb = Kh + ((size_t)(b*KV + kvh)*SEQ)*DHEAD;
  const __bf16* Vb = Vt + ((size_t)(b*KV + kvh)*DHEAD)*SEQ;
  const int kbmax = (blockIdx.x*128 + 127) >> 5;  // causal upper key block (block-uniform)
  const int myqmax = q0 + 15;
  for (int kb = 0; kb <= kbmax; ++kb) {
    const int t0 = kb*32;
    // stage K (32x256) and Vt (256x32) tiles; bottom-of-loop barrier protects reuse
    #pragma unroll
    for (int i = 0; i < 4; ++i) {
      int c = tid + i*256;
      int rowk = c >> 5, colk = (c & 31)*8;
      async_cp16(Kb + (size_t)(t0 + rowk)*DHEAD + colk, &Ks[rowk*KP + colk]);
      int rowv = c >> 2, colv = (c & 3)*8;
      async_cp16(Vb + (size_t)rowv*SEQ + t0 + colv, &Vs[rowv*VP + colv]);
    }
    async_wait0();
    __syncthreads();
    if (t0 <= myqmax) {
      v8f s0 = {}, s1 = {};
      #pragma unroll
      for (int kk = 0; kk < 8; ++kk) {            // S = Q K^T over D=256
        v16bf bk0 = ldfragB(&Ks[lm*KP + kk*32], half);
        v16bf bk1 = ldfragB(&Ks[(16 + lm)*KP + kk*32], half);
        s0 = wmma_bf16(qf[kk], bk0, s0);
        s1 = wmma_bf16(qf[kk], bk1, s1);
      }
      float alpha[8];
      __bf16* pw = &Ps[wave*16*VP];
      #pragma unroll
      for (int r = 0; r < 8; ++r) {               // online softmax stats per row
        const int qrow = q0 + r + 8*half;
        float x0 = s0[r]*ATT_SCALE; if (t0 + lm > qrow)      x0 = -1e30f;
        float x1 = s1[r]*ATT_SCALE; if (t0 + 16 + lm > qrow) x1 = -1e30f;
        float mx = fmaxf(x0, x1);
        mx = fmaxf(mx, __shfl_xor(mx, 1, 32));
        mx = fmaxf(mx, __shfl_xor(mx, 2, 32));
        mx = fmaxf(mx, __shfl_xor(mx, 4, 32));
        mx = fmaxf(mx, __shfl_xor(mx, 8, 32));
        const float mnew = fmaxf(mst[r], mx);
        const float a  = __expf(mst[r] - mnew);
        const float p0 = __expf(x0 - mnew);
        const float p1 = __expf(x1 - mnew);
        float psum = p0 + p1;
        psum += __shfl_xor(psum, 1, 32);
        psum += __shfl_xor(psum, 2, 32);
        psum += __shfl_xor(psum, 4, 32);
        psum += __shfl_xor(psum, 8, 32);
        lst[r] = lst[r]*a + psum;
        mst[r] = mnew;
        alpha[r] = a;
        __bf16* prow = pw + (r + 8*half)*VP;      // C-layout -> LDS (16x32)
        prow[lm]      = (__bf16)p0;
        prow[16 + lm] = (__bf16)p1;
      }
      #pragma unroll
      for (int nd = 0; nd < 16; ++nd)
        #pragma unroll
        for (int r = 0; r < 8; ++r)
          o[nd][r] *= alpha[r];
      asm volatile("s_wait_dscnt 0x0" ::: "memory"); // P stores visible before A-layout reload
      v16bf pf = ldfragA(pw + lm*VP, half);
      #pragma unroll
      for (int nd = 0; nd < 16; ++nd) {           // O += P V over 32 keys, D=256
        v16bf vf = ldfragB(&Vs[(nd*16 + lm)*VP], half);
        o[nd] = wmma_bf16(pf, vf, o[nd]);
      }
    }
    __syncthreads();
  }
  float inv[8];
  #pragma unroll
  for (int r = 0; r < 8; ++r) inv[r] = 1.f / lst[r];
  __bf16* Ob = Obf + ((size_t)(b*SEQ + q0))*NQ + h*DHEAD;
  #pragma unroll
  for (int nd = 0; nd < 16; ++nd)
    #pragma unroll
    for (int r = 0; r < 8; ++r)
      Ob[(size_t)(r + 8*half)*NQ + nd*16 + lm] = (__bf16)(o[nd][r]*inv[r]);
}

// ---------------- launch ----------------
extern "C" void kernel_launch(void* const* d_in, const int* in_sizes, int n_in,
                              void* d_out, int out_size, void* d_ws, size_t ws_size,
                              hipStream_t stream) {
  (void)in_sizes; (void)n_in; (void)out_size; (void)ws_size;
  const float* x  = (const float*)d_in[0];
  const float* wq = (const float*)d_in[1];
  const float* wk = (const float*)d_in[2];
  const float* wv = (const float*)d_in[3];
  const float* wo = (const float*)d_in[4];
  const float* qw = (const float*)d_in[5];
  const float* kw = (const float*)d_in[6];
  // d_in[7] = attention_mask (pure causal; applied analytically in-kernel)
  const int* pos  = (const int*)d_in[8];
  float* out = (float*)d_out;

  char* ws = (char*)d_ws;
  size_t off = 0;
  __bf16* x_bf    = (__bf16*)(ws + off); off += (size_t)MROWS*HIDDEN*2;
  __bf16* wqkvT   = (__bf16*)(ws + off); off += (size_t)NQKV*HIDDEN*2;
  __bf16* woT     = (__bf16*)(ws + off); off += (size_t)HIDDEN*NQ*2;
  float*  qkv_raw = (float*) (ws + off); off += (size_t)MROWS*NQKV*4;
  __bf16* Qh      = (__bf16*)(ws + off); off += (size_t)BATCH*H*SEQ*DHEAD*2;
  __bf16* Kh      = (__bf16*)(ws + off); off += (size_t)BATCH*KV*SEQ*DHEAD*2;
  __bf16* Vt      = (__bf16*)(ws + off); off += (size_t)BATCH*KV*SEQ*DHEAD*2;
  __bf16* Obf     = (__bf16*)(ws + off); off += (size_t)MROWS*NQ*2;

  cvt_f32_bf16_kernel<<<(MROWS*HIDDEN)/256, 256, 0, stream>>>(x, x_bf, MROWS*HIDDEN);
  build_wqkvT_kernel<<<(NQKV*HIDDEN)/256, 256, 0, stream>>>(wq, wk, wv, wqkvT);
  build_woT_kernel<<<(HIDDEN*NQ)/256, 256, 0, stream>>>(wo, woT);
  gemm_bf16_kernel<<<dim3(NQKV/128, MROWS/128), 256, 0, stream>>>(
      x_bf, wqkvT, qkv_raw, HIDDEN, HIDDEN, HIDDEN, NQKV);
  rmsrope_kernel<<<MROWS, 384, 0, stream>>>(qkv_raw, qw, kw, pos, Qh, Kh);
  vtrans_kernel<<<(BATCH*KV*DHEAD*SEQ)/256, 256, 0, stream>>>(qkv_raw, Vt);
  attn_kernel<<<dim3(SEQ/128, H, BATCH), 256, 0, stream>>>(Qh, Kh, Vt, Obf);
  gemm_bf16_kernel<<<dim3(HIDDEN/128, MROWS/128), 256, 0, stream>>>(
      Obf, woT, out, NQ, NQ, NQ, HIDDEN);
}
